// AutoregressiveRNN_67860483277101
// MI455X (gfx1250) — matrix operand order repaired and checked by reading, not compiled, not run on hardware
//
#include <hip/hip_runtime.h>

// GRU: B=64, T=1024, C=128, H=512, D=512
#define BB 64
#define TT 1024
#define CC 128
#define HH 512
#define DD 512
#define BT (BB * TT)     // 65536
#define G3 (3 * HH)      // 1536

typedef __attribute__((ext_vector_type(16))) __bf16         v16bf;
typedef __attribute__((ext_vector_type(8)))  float          v8f;
typedef __attribute__((ext_vector_type(16))) unsigned short v16u;
typedef __attribute__((ext_vector_type(8)))  unsigned short v8u;

__device__ __forceinline__ unsigned short f2bf(float f) {
  unsigned int u = __float_as_uint(f);
  u += 0x7FFFu + ((u >> 16) & 1u);     // round-to-nearest-even
  return (unsigned short)(u >> 16);
}

__device__ __forceinline__ v8f wmma_bf16(v16bf a, v16bf b, v8f c) {
  return __builtin_amdgcn_wmma_f32_16x16x32_bf16(false, a, false, b, (short)0, c,
                                                 false, false);
}

// Async global->LDS copy, 16 bytes per lane; tracked by ASYNCcnt.
// VDST = VGPR with wave-relative LDS byte offset, VADDR = 64-bit global address.
__device__ __forceinline__ void async_copy_b128(unsigned lds_off, const void* g) {
  asm volatile("global_load_async_to_lds_b128 %0, %1, off"
               :: "v"(lds_off), "v"((unsigned long long)(size_t)g)
               : "memory");
}
__device__ __forceinline__ void wait_async0() {
  asm volatile("s_wait_asynccnt 0" ::: "memory");
}

// A-fragment (16x32 bf16, MxK), row-major source, base = tile row 0, lead dim ld.
__device__ __forceinline__ v16bf load_a_frag(const unsigned short* base, int ld,
                                             int lane, int k0) {
  const int m  = lane & 15;
  const int h8 = (lane >> 4) << 3;
  const v8u lo = *(const v8u*)(base + m * ld + k0 + h8);
  const v8u hi = *(const v8u*)(base + m * ld + k0 + 16 + h8);
  v16u r;
#pragma unroll
  for (int i = 0; i < 8; ++i) { r[i] = lo[i]; r[i + 8] = hi[i]; }
  return __builtin_bit_cast(v16bf, r);
}

// B-fragment (32x16 bf16, KxN): B[k][n] = W[g0+n][k0+k], W row-major lead dim ld.
__device__ __forceinline__ v16bf load_b_frag(const unsigned short* W, int ld,
                                             int lane, int g0, int k0) {
  const int n   = lane & 15;
  const int h8  = (lane >> 4) << 3;
  const v8u lo  = *(const v8u*)(W + (size_t)(g0 + n) * ld + k0 + h8);
  const v8u hi  = *(const v8u*)(W + (size_t)(g0 + n) * ld + k0 + 16 + h8);
  v16u v;
#pragma unroll
  for (int i = 0; i < 8; ++i) { v[i] = lo[i]; v[i + 8] = hi[i]; }
  return __builtin_bit_cast(v16bf, v);
}

// ---------------------------------------------------------------- prep: f32->bf16
__global__ __launch_bounds__(256) void k_cvt_bf16(const float* __restrict__ in,
                                                  unsigned short* __restrict__ out,
                                                  int n) {
  int i = blockIdx.x * blockDim.x + threadIdx.x;
  if (i < n) out[i] = f2bf(in[i]);
}

// ------------------------------------------- phase 1: x_gates = x @ w_ih^T + b_ih
// M=BT, N=G3, K=CC. One wave computes a 32x64 tile: 2 M-tiles x 4 N-tiles.
__global__ __launch_bounds__(256) void k_xgates(
    const unsigned short* __restrict__ x_bf,    // [BT][CC]
    const unsigned short* __restrict__ wih_bf,  // [G3][CC]
    const float* __restrict__ b_ih,             // [G3]
    float* __restrict__ xg)                     // [BT][G3]
{
  const int lane = threadIdx.x & 31;
  const int wave = threadIdx.x >> 5;
  const int m0 = blockIdx.x * 32;
  const int g0 = blockIdx.y * 512 + wave * 64;
  const unsigned short* a0base = x_bf + (size_t)m0 * CC;
  const unsigned short* a1base = a0base + 16 * CC;

  v8f acc[2][4] = {};
#pragma unroll
  for (int k0 = 0; k0 < CC; k0 += 32) {
    v16bf a0 = load_a_frag(a0base, CC, lane, k0);
    v16bf a1 = load_a_frag(a1base, CC, lane, k0);
#pragma unroll
    for (int t = 0; t < 4; ++t) {
      v16bf b = load_b_frag(wih_bf, CC, lane, g0 + t * 16, k0);
      acc[0][t] = wmma_bf16(a0, b, acc[0][t]);
      acc[1][t] = wmma_bf16(a1, b, acc[1][t]);
    }
  }
  const int n = lane & 15;
#pragma unroll
  for (int mt = 0; mt < 2; ++mt) {
    const int mrow = m0 + mt * 16 + ((lane >> 4) << 3);
#pragma unroll
    for (int t = 0; t < 4; ++t) {
      int g = g0 + t * 16 + n;
      float bias = b_ih[g];
#pragma unroll
      for (int v = 0; v < 8; ++v)
        xg[(size_t)(mrow + v) * G3 + g] = acc[mt][t][v] + bias;
    }
  }
}

// --------------------------------------------------- phase 2: sequential GRU scan
// 4 blocks (16 batch rows each) x 32 waves. Wave w owns gate columns 16w..16w+15
// of r, z, n. h in LDS (f32 + bf16); per-step xg slice staged into LDS with
// GLOBAL_LOAD_ASYNC_TO_LDS_B128, overlapped with the WMMA loop; WMMA operand
// streams are register double-buffered.
__global__ __launch_bounds__(1024) void k_scan(
    const float* __restrict__ h0,               // [BB][HH]
    const unsigned short* __restrict__ whh_bf,  // [G3][HH]
    const float* __restrict__ b_hh,             // [G3]
    const float* __restrict__ xg,               // [BB][TT][G3]
    unsigned short* __restrict__ hs_bf,         // [BB][TT][HH]
    float* __restrict__ hfinal)                 // [BB][HH]
{
  __shared__ float          h_f32[16 * HH];     // 32 KB
  __shared__ unsigned short h_b16[16 * HH];     // 16 KB
  __shared__ float          xbuf[16 * G3];      // 96 KB  (one timestep's gates)

  const int tid  = threadIdx.x;
  const int lane = tid & 31;
  const int wave = tid >> 5;
  const int b0   = blockIdx.x * 16;

  for (int i = tid; i < 16 * HH; i += 1024) {
    float v = h0[(size_t)(b0 + (i >> 9)) * HH + (i & (HH - 1))];
    h_f32[i] = v;
    h_b16[i] = f2bf(v);
  }

  // Per-thread async-copy slots: 6 x 16B chunks cover 16*G3 floats with 1024 thr.
  unsigned     cp_lds[6];
  const float* cp_src[6];
#pragma unroll
  for (int c = 0; c < 6; ++c) {
    int e   = (c * 1024 + tid) * 4;             // float index within slice
    int row = e / G3;
    int col = e - row * G3;
    cp_lds[c] = (unsigned)(size_t)&xbuf[e];
    cp_src[c] = xg + (size_t)(b0 + row) * TT * G3 + col;
  }

  const int j0 = wave * 16;                     // column tile base within H
  const int n  = lane & 15;
  const int mh = (lane >> 4) << 3;              // 0 or 8
  const int j  = j0 + n;
  const float bhr = b_hh[j];
  const float bhz = b_hh[HH + j];
  const float bhn = b_hh[2 * HH + j];

  for (int t = 0; t < TT; ++t) {
    __syncthreads();   // sync1: new h visible; xbuf free (prev step's reads done)

    // Kick off async staging of this step's gate pre-activations into LDS.
#pragma unroll
    for (int c = 0; c < 6; ++c)
      async_copy_b128(cp_lds[c], cp_src[c] + (size_t)t * G3);

    // hg = h @ w_hh^T : 16 k-chunks, register double-buffered.
    v8f ar = {}, az = {}, an = {};
    v16bf a0 = load_a_frag(h_b16, HH, lane, 0);
    v16bf r0 = load_b_frag(whh_bf, HH, lane, j0, 0);
    v16bf z0 = load_b_frag(whh_bf, HH, lane, HH + j0, 0);
    v16bf n0 = load_b_frag(whh_bf, HH, lane, 2 * HH + j0, 0);
    for (int k0 = 0; k0 < HH - 32; k0 += 32) {
      const int kn = k0 + 32;
      v16bf a1 = load_a_frag(h_b16, HH, lane, kn);
      v16bf r1 = load_b_frag(whh_bf, HH, lane, j0, kn);
      v16bf z1 = load_b_frag(whh_bf, HH, lane, HH + j0, kn);
      v16bf n1 = load_b_frag(whh_bf, HH, lane, 2 * HH + j0, kn);
      ar = wmma_bf16(a0, r0, ar);
      az = wmma_bf16(a0, z0, az);
      an = wmma_bf16(a0, n0, an);
      a0 = a1; r0 = r1; z0 = z1; n0 = n1;
    }
    ar = wmma_bf16(a0, r0, ar);
    az = wmma_bf16(a0, z0, az);
    an = wmma_bf16(a0, n0, an);

    wait_async0();     // my async LDS writes complete
    __syncthreads();   // sync2: whole xg slice visible in LDS

    float hnew[8];
#pragma unroll
    for (int v = 0; v < 8; ++v) {
      int row = mh + v;                         // 0..15
      float xr = xbuf[row * G3 + j];
      float xz = xbuf[row * G3 + HH + j];
      float xn = xbuf[row * G3 + 2 * HH + j];
      float ho = h_f32[row * HH + j];
      float r  = 1.0f / (1.0f + __expf(-(xr + ar[v] + bhr)));
      float z  = 1.0f / (1.0f + __expf(-(xz + az[v] + bhz)));
      float nn = tanhf(xn + r * (an[v] + bhn));
      hnew[v] = (1.0f - z) * nn + z * ho;
    }
    __syncthreads();   // sync3: all reads of old h / xbuf done
#pragma unroll
    for (int v = 0; v < 8; ++v) {
      int row = mh + v;
      unsigned short hb = f2bf(hnew[v]);
      h_f32[row * HH + j] = hnew[v];
      h_b16[row * HH + j] = hb;
      hs_bf[((size_t)(b0 + row) * TT + t) * HH + j] = hb;
    }
  }
  __syncthreads();
  for (int i = tid; i < 16 * HH; i += 1024)
    hfinal[(size_t)(b0 + (i >> 9)) * HH + (i & (HH - 1))] = h_f32[i];
}

// ------------------------------------ phase 3: out = relu(hs @ w_post^T + b_post)
// M=BT, N=DD, K=HH. One wave computes a 32x64 tile: 2 M-tiles x 4 N-tiles.
__global__ __launch_bounds__(256) void k_post(
    const unsigned short* __restrict__ hs_bf,    // [BT][HH]
    const unsigned short* __restrict__ wpost_bf, // [DD][HH]
    const float* __restrict__ b_post,            // [DD]
    float* __restrict__ out)                     // [BT][DD]
{
  const int lane = threadIdx.x & 31;
  const int wave = threadIdx.x >> 5;
  const int m0 = blockIdx.x * 32;
  const int g0 = wave * 64;
  const unsigned short* a0base = hs_bf + (size_t)m0 * HH;
  const unsigned short* a1base = a0base + 16 * HH;

  v8f acc[2][4] = {};
#pragma unroll 4
  for (int k0 = 0; k0 < HH; k0 += 32) {
    v16bf a0 = load_a_frag(a0base, HH, lane, k0);
    v16bf a1 = load_a_frag(a1base, HH, lane, k0);
#pragma unroll
    for (int t = 0; t < 4; ++t) {
      v16bf b = load_b_frag(wpost_bf, HH, lane, g0 + t * 16, k0);
      acc[0][t] = wmma_bf16(a0, b, acc[0][t]);
      acc[1][t] = wmma_bf16(a1, b, acc[1][t]);
    }
  }
  const int n = lane & 15;
#pragma unroll
  for (int mt = 0; mt < 2; ++mt) {
    const int mrow = m0 + mt * 16 + ((lane >> 4) << 3);
#pragma unroll
    for (int t = 0; t < 4; ++t) {
      int g = g0 + t * 16 + n;
      float bias = b_post[g];
#pragma unroll
      for (int v = 0; v < 8; ++v) {
        float val = acc[mt][t][v] + bias;
        out[(size_t)(mrow + v) * DD + g] = val > 0.0f ? val : 0.0f;
      }
    }
  }
}

// -------------------------------------------------------------------------- launch
extern "C" void kernel_launch(void* const* d_in, const int* in_sizes, int n_in,
                              void* d_out, int out_size, void* d_ws, size_t ws_size,
                              hipStream_t stream) {
  const float* x      = (const float*)d_in[0];   // [B,T,C]
  const float* h0     = (const float*)d_in[1];   // [B,H]
  const float* w_ih   = (const float*)d_in[2];   // [3H,C]
  const float* w_hh   = (const float*)d_in[3];   // [3H,H]
  const float* b_ih   = (const float*)d_in[4];   // [3H]
  const float* b_hh   = (const float*)d_in[5];   // [3H]
  const float* w_post = (const float*)d_in[6];   // [D,H]
  const float* b_post = (const float*)d_in[7];   // [D]
  float* out = (float*)d_out;                    // [B,T,D] then [B,H]

  char* ws = (char*)d_ws;
  size_t off = 0;
  auto carve = [&](size_t bytes) {
    char* p = ws + off;
    off = (off + bytes + 255) & ~(size_t)255;
    return p;
  };
  unsigned short* x_bf     = (unsigned short*)carve((size_t)BT * CC * 2);
  unsigned short* wih_bf   = (unsigned short*)carve((size_t)G3 * CC * 2);
  unsigned short* whh_bf   = (unsigned short*)carve((size_t)G3 * HH * 2);
  unsigned short* wpost_bf = (unsigned short*)carve((size_t)DD * HH * 2);
  float*          xg       = (float*)carve((size_t)BT * G3 * 4);
  unsigned short* hs_bf    = (unsigned short*)carve((size_t)BT * HH * 2);
  (void)ws_size; (void)in_sizes; (void)n_in; (void)out_size;

  {
    int n;
    n = BT * CC;  k_cvt_bf16<<<(n + 255) / 256, 256, 0, stream>>>(x,      x_bf,     n);
    n = G3 * CC;  k_cvt_bf16<<<(n + 255) / 256, 256, 0, stream>>>(w_ih,   wih_bf,   n);
    n = G3 * HH;  k_cvt_bf16<<<(n + 255) / 256, 256, 0, stream>>>(w_hh,   whh_bf,   n);
    n = DD * HH;  k_cvt_bf16<<<(n + 255) / 256, 256, 0, stream>>>(w_post, wpost_bf, n);
  }

  // phase 1: x_gates  (2048 M-tiles x 3 N-slabs of 512)
  k_xgates<<<dim3(BT / 32, G3 / 512), 256, 0, stream>>>(x_bf, wih_bf, b_ih, xg);

  // phase 2: sequential scan; h_final -> tail of d_out
  k_scan<<<BB / 16, 1024, 0, stream>>>(h0, whh_bf, b_hh, xg, hs_bf,
                                       out + (size_t)BT * DD);

  // phase 3: post projection + ReLU
  k_post<<<BT / 32, 256, 0, stream>>>(hs_bf, wpost_bf, b_post, out);
}
